// PointNetPlusPlusAttnFusion_20727512170872
// MI455X (gfx1250) — compile-verified, matrix operand order checked by reading, not run on hardware
//
#include <hip/hip_runtime.h>
#include <hip/hip_bf16.h>

typedef __attribute__((ext_vector_type(16))) _Float16 v16h;
typedef __attribute__((ext_vector_type(8)))  float    v8f;

#define DEV __device__ __forceinline__

// ---------------------------------------------------------------------------
// WMMA helpers (CDNA5 wave32, V_WMMA_F32_16X16X32_F16)
// A fragment: lane L -> row = row0 + (L&15), half = L>>4,
//   element e -> k = kbase + (e<8?0:16) + half*8 + (e%8)
// B fragment: same loader over a [N][Kpad] row-major array (column-major B).
// C/D: lane L -> col = L&15, vgpr v -> row = v + 8*(L>>4).
// ---------------------------------------------------------------------------
DEV v16h load_frag(const _Float16* __restrict__ p0, int row0, int kbase, int ldk,
                   int lane) {
  const _Float16* p = p0 + (size_t)(row0 + (lane & 15)) * ldk + kbase + ((lane >> 4) << 3);
  v16h f;
#pragma unroll
  for (int e = 0; e < 8; ++e) f[e] = p[e];
#pragma unroll
  for (int e = 0; e < 8; ++e) f[e + 8] = p[e + 16];
  return f;
}

DEV v8f wmma32(v16h a, v16h b, v8f c) {
  return __builtin_amdgcn_wmma_f32_16x16x32_f16(false, a, false, b, (short)0, c,
                                                false, false);
}

// Async copy of a contiguous tile (global -> LDS), 512B per instruction per
// wave (16B/lane), tracked by ASYNCcnt. gfx1250 async-DMA path.
DEV void async_stage(const _Float16* g, _Float16* lds, int bytes, int lane) {
  unsigned long long ga = (unsigned long long)g + (unsigned)(lane * 16);
  unsigned la = (unsigned)(unsigned long long)(void*)lds + (unsigned)(lane * 16);
  for (int i = 0; i < bytes; i += 512) {
    asm volatile("global_load_async_to_lds_b128 %0, %1, off"
                 :: "v"(la), "v"(ga) : "memory");
    ga += 512;
    la += 512;
  }
  asm volatile("s_wait_asynccnt 0" ::: "memory");
}

// out[m][n] = relu(sum_k in[m][k]*w[n][k] + bias[n]),   M = MT*16 rows
DEV void mlp_layer(const _Float16* __restrict__ in, int ldin,
                   const _Float16* __restrict__ w, const float* __restrict__ bias,
                   int Kpad, int N, int MT, _Float16* __restrict__ out, int ldout,
                   int lane) {
  for (int nt = 0; nt < N; nt += 16) {
    for (int mt = 0; mt < MT; ++mt) {
      v8f acc = {0.f, 0.f, 0.f, 0.f, 0.f, 0.f, 0.f, 0.f};
      for (int kt = 0; kt < Kpad; kt += 32) {
        v16h a  = load_frag(in, mt * 16, kt, ldin, lane);
        v16h bb = load_frag(w, nt, kt, Kpad, lane);
        acc = wmma32(a, bb, acc);
      }
      float bn  = bias[nt + (lane & 15)];
      int mbase = mt * 16 + ((lane >> 4) << 3);
      int ncol  = nt + (lane & 15);
#pragma unroll
      for (int v = 0; v < 8; ++v) {
        float r = fmaxf(acc[v] + bn, 0.f);
        out[(size_t)(mbase + v) * ldout + ncol] = (_Float16)r;
      }
    }
  }
}

// same but f32 store to two destinations (final layer writes d_out twice)
DEV void mlp_layer_f32dual(const _Float16* __restrict__ in, int ldin,
                           const _Float16* __restrict__ w,
                           const float* __restrict__ bias, int Kpad, int N, int MT,
                           float* __restrict__ oA, float* __restrict__ oB,
                           int ldout, int lane) {
  for (int nt = 0; nt < N; nt += 16) {
    for (int mt = 0; mt < MT; ++mt) {
      v8f acc = {0.f, 0.f, 0.f, 0.f, 0.f, 0.f, 0.f, 0.f};
      for (int kt = 0; kt < Kpad; kt += 32) {
        v16h a  = load_frag(in, mt * 16, kt, ldin, lane);
        v16h bb = load_frag(w, nt, kt, Kpad, lane);
        acc = wmma32(a, bb, acc);
      }
      float bn  = bias[nt + (lane & 15)];
      int mbase = mt * 16 + ((lane >> 4) << 3);
      int ncol  = nt + (lane & 15);
#pragma unroll
      for (int v = 0; v < 8; ++v) {
        float r = fmaxf(acc[v] + bn, 0.f);
        size_t o = (size_t)(mbase + v) * ldout + ncol;
        oA[o] = r;
        oB[o] = r;
      }
    }
  }
}

// relu(max over all M rows) per output channel (max(relu(x)) == relu(max(x)))
template <typename T>
DEV void mlp_layer_max(const _Float16* __restrict__ in, int ldin,
                       const _Float16* __restrict__ w,
                       const float* __restrict__ bias, int Kpad, int N, int MT,
                       T* __restrict__ outmax, int lane) {
  for (int nt = 0; nt < N; nt += 16) {
    float m  = -1e30f;
    float bn = bias[nt + (lane & 15)];
    for (int mt = 0; mt < MT; ++mt) {
      v8f acc = {0.f, 0.f, 0.f, 0.f, 0.f, 0.f, 0.f, 0.f};
      for (int kt = 0; kt < Kpad; kt += 32) {
        v16h a  = load_frag(in, mt * 16, kt, ldin, lane);
        v16h bb = load_frag(w, nt, kt, Kpad, lane);
        acc = wmma32(a, bb, acc);
      }
#pragma unroll
      for (int v = 0; v < 8; ++v) m = fmaxf(m, acc[v] + bn);
    }
    m = fmaxf(m, __shfl_xor(m, 16, 32));  // combine the two 16-lane halves
    m = fmaxf(m, 0.f);
    if (lane < 16) outmax[nt + lane] = (T)m;
  }
}

// ---------------------------------------------------------------------------
// Weight conversion: f32 (N,K) row-major -> f16 (N,Kpad), zero padded
// ---------------------------------------------------------------------------
__global__ void cvt_w(const float* __restrict__ src, _Float16* __restrict__ dst,
                      int N, int K, int Kpad) {
  int t = blockIdx.x * blockDim.x + threadIdx.x;
  if (t >= N * Kpad) return;
  int n = t / Kpad, k = t - n * Kpad;
  dst[t] = (k < K) ? (_Float16)src[(size_t)n * K + k] : (_Float16)0.f;
}

__global__ void copy_f32(const float* __restrict__ s, float* __restrict__ d, int n) {
  int t = blockIdx.x * blockDim.x + threadIdx.x;
  if (t < n) d[t] = s[t];
}

// ---------------------------------------------------------------------------
// Farthest point sampling: one 512-thread block per batch. NPT is a template
// parameter so the per-thread distance cache stays in VGPRs (no scratch in the
// only serial loop of the whole network).
// ---------------------------------------------------------------------------
template <int NPT>
__global__ void __launch_bounds__(512) fps_kernel(const float* __restrict__ xyz,
                                                  int N, int S,
                                                  int* __restrict__ fps_idx,
                                                  float* __restrict__ new_xyz) {
  __shared__ float sval[512];
  __shared__ int   sidx[512];
  __shared__ int   curfar;
  int b = blockIdx.x, t = threadIdx.x;
  const float* P = xyz + (size_t)b * N * 3;
  float d[NPT];
#pragma unroll
  for (int i = 0; i < NPT; ++i) d[i] = 1e10f;
  if (t == 0) curfar = 0;
  __syncthreads();
  for (int it = 0; it < S; ++it) {
    int far = curfar;
    if (t == 0) {
      fps_idx[(size_t)b * S + it] = far;
      new_xyz[((size_t)b * S + it) * 3 + 0] = P[far * 3 + 0];
      new_xyz[((size_t)b * S + it) * 3 + 1] = P[far * 3 + 1];
      new_xyz[((size_t)b * S + it) * 3 + 2] = P[far * 3 + 2];
    }
    float cx = P[far * 3 + 0], cy = P[far * 3 + 1], cz = P[far * 3 + 2];
    float best = -1.f;
    int bi = 0;
#pragma unroll
    for (int i = 0; i < NPT; ++i) {
      int pi = t * NPT + i;
      float dx = P[pi * 3 + 0] - cx, dy = P[pi * 3 + 1] - cy, dz = P[pi * 3 + 2] - cz;
      float dd = dx * dx + dy * dy + dz * dz;
      d[i] = fminf(d[i], dd);
      if (d[i] > best) { best = d[i]; bi = pi; }
    }
    sval[t] = best; sidx[t] = bi;
    __syncthreads();
    for (int off = 256; off > 0; off >>= 1) {
      if (t < off) {
        if (sval[t + off] > sval[t] ||
            (sval[t + off] == sval[t] && sidx[t + off] < sidx[t])) {
          sval[t] = sval[t + off]; sidx[t] = sidx[t + off];
        }
      }
      __syncthreads();
    }
    if (t == 0) curfar = sidx[0];
    __syncthreads();
  }
}

// ---------------------------------------------------------------------------
// Ball query: one wave per centroid, ballot+prefix append of in-radius indices
// in ascending order (matches the reference sort-by-index semantics).
// ---------------------------------------------------------------------------
__global__ void __launch_bounds__(32) ball_query_kernel(
    const float* __restrict__ xyz, const float* __restrict__ new_xyz,
    int N, int S, float r2, int K, int* __restrict__ out) {
  int g = blockIdx.x;
  int b = g / S, s = g - b * S;
  int lane = threadIdx.x;
  const float* P = xyz + (size_t)b * N * 3;
  const float* q = new_xyz + ((size_t)b * S + s) * 3;
  float qx = q[0], qy = q[1], qz = q[2];
  int* o = out + (size_t)g * K;
  int cnt = 0;
  for (int base = 0; base < N && cnt < K; base += 32) {
    int pi = base + lane;
    float dx = P[pi * 3 + 0] - qx, dy = P[pi * 3 + 1] - qy, dz = P[pi * 3 + 2] - qz;
    float dd = dx * dx + dy * dy + dz * dz;
    bool inr = dd <= r2;
    unsigned mask = (unsigned)__ballot(inr);
    int pre = __popc(mask & ((1u << lane) - 1u));
    if (inr && (cnt + pre) < K) o[cnt + pre] = pi;
    cnt += __popc(mask);
  }
  if (cnt > K) cnt = K;
  int first = o[0];  // same-wave store->load on same address is ordered
  for (int j = cnt + lane; j < K; j += 32) o[j] = first;
}

// ---------------------------------------------------------------------------
// SA1: group (32 samples x 6ch -> pad32) -> 64 -> 64 -> 128 -> max
// ---------------------------------------------------------------------------
__global__ void __launch_bounds__(32) sa1_kernel(
    const float* __restrict__ xyz, const float* __restrict__ new_xyz,
    const int* __restrict__ ball,
    const _Float16* w1, const float* b1, const _Float16* w2, const float* b2,
    const _Float16* w3, const float* b3, _Float16* __restrict__ out) {
  __shared__ _Float16 bufA[32 * 64];
  __shared__ _Float16 bufB[32 * 64];
  int g = blockIdx.x;
  int b = g >> 9, s = g & 511;
  int lane = threadIdx.x;
  const float* c = new_xyz + ((size_t)b * 512 + s) * 3;
  float cx = c[0], cy = c[1], cz = c[2];
  int id = ball[(size_t)g * 32 + lane];
  const float* p = xyz + ((size_t)b * 8192 + id) * 3;
  _Float16* row = bufA + lane * 32;
  row[0] = (_Float16)(p[0] - cx); row[1] = (_Float16)(p[1] - cy);
  row[2] = (_Float16)(p[2] - cz);
  row[3] = (_Float16)p[0]; row[4] = (_Float16)p[1]; row[5] = (_Float16)p[2];
  for (int k = 6; k < 32; ++k) row[k] = (_Float16)0.f;
  __syncthreads();
  mlp_layer(bufA, 32, w1, b1, 32, 64, 2, bufB, 64, lane);
  __syncthreads();
  mlp_layer(bufB, 64, w2, b2, 64, 64, 2, bufA, 64, lane);
  __syncthreads();
  mlp_layer_max(bufA, 64, w3, b3, 64, 128, 2, out + (size_t)g * 128, lane);
}

// ---------------------------------------------------------------------------
// SA2: group (64 samples x 131ch -> pad160) -> 128 -> 128 -> 256 -> max
// ---------------------------------------------------------------------------
__global__ void __launch_bounds__(32) sa2_kernel(
    const float* __restrict__ l1xyz, const float* __restrict__ new_xyz,
    const int* __restrict__ ball, const _Float16* __restrict__ l1pts,
    const _Float16* w1, const float* b1, const _Float16* w2, const float* b2,
    const _Float16* w3, const float* b3, _Float16* __restrict__ out) {
  __shared__ _Float16 bufA[64 * 160];
  __shared__ _Float16 bufB[64 * 128];
  int g = blockIdx.x;
  int b = g >> 7, s = g & 127;
  int lane = threadIdx.x;
  const float* c = new_xyz + ((size_t)b * 128 + s) * 3;
  float cx = c[0], cy = c[1], cz = c[2];
  for (int r = lane; r < 64; r += 32) {
    int id = ball[(size_t)g * 64 + r];
    const float* p = l1xyz + ((size_t)b * 512 + id) * 3;
    const _Float16* f = l1pts + ((size_t)b * 512 + id) * 128;
    _Float16* row = bufA + r * 160;
    row[0] = (_Float16)(p[0] - cx); row[1] = (_Float16)(p[1] - cy);
    row[2] = (_Float16)(p[2] - cz);
    for (int k = 0; k < 128; ++k) row[3 + k] = f[k];
    for (int k = 131; k < 160; ++k) row[k] = (_Float16)0.f;
  }
  __syncthreads();
  mlp_layer(bufA, 160, w1, b1, 160, 128, 4, bufB, 128, lane);
  __syncthreads();
  mlp_layer(bufB, 128, w2, b2, 128, 128, 4, bufA, 128, lane);
  __syncthreads();
  mlp_layer_max(bufA, 128, w3, b3, 128, 256, 4, out + (size_t)g * 256, lane);
}

// ---------------------------------------------------------------------------
// SA3 (global): 128 rows split into 4 chunks of 32; 259ch(pad288)->256->512->1024,
// partial max per chunk then combined.
// ---------------------------------------------------------------------------
__global__ void __launch_bounds__(32) sa3_kernel(
    const float* __restrict__ l2xyz, const _Float16* __restrict__ l2pts,
    const _Float16* w1, const float* b1, const _Float16* w2, const float* b2,
    const _Float16* w3, const float* b3, float* __restrict__ partial) {
  __shared__ _Float16 bufA[32 * 512];
  __shared__ _Float16 bufB[32 * 256];
  int g = blockIdx.x;           // B*4
  int b = g >> 2, chunk = g & 3;
  int lane = threadIdx.x;
  int pidx = chunk * 32 + lane;
  const float* q = l2xyz + ((size_t)b * 128 + pidx) * 3;
  const _Float16* f = l2pts + ((size_t)b * 128 + pidx) * 256;
  _Float16* row = bufA + lane * 288;
  row[0] = (_Float16)q[0]; row[1] = (_Float16)q[1]; row[2] = (_Float16)q[2];
  for (int k = 0; k < 256; ++k) row[3 + k] = f[k];
  for (int k = 259; k < 288; ++k) row[k] = (_Float16)0.f;
  __syncthreads();
  mlp_layer(bufA, 288, w1, b1, 288, 256, 2, bufB, 256, lane);
  __syncthreads();
  mlp_layer(bufB, 256, w2, b2, 256, 512, 2, bufA, 512, lane);
  __syncthreads();
  mlp_layer_max(bufA, 512, w3, b3, 512, 1024, 2, partial + (size_t)g * 1024, lane);
}

__global__ void sa3_combine(const float* __restrict__ partial, float* __restrict__ l3) {
  int t = blockIdx.x * blockDim.x + threadIdx.x;
  if (t >= 16 * 1024) return;
  int b = t >> 10, n = t & 1023;
  const float* p = partial + (size_t)b * 4 * 1024 + n;
  float m = fmaxf(fmaxf(p[0], p[1024]), fmaxf(p[2048], p[3072]));
  l3[t] = m;
}

// ---------------------------------------------------------------------------
// FP3 staging: interp is a broadcast of l3 (S==1); concat [l2_pts(256), l3(1024)]
// ---------------------------------------------------------------------------
__global__ void fp3_stage(const _Float16* __restrict__ l2pts,
                          const float* __restrict__ l3,
                          _Float16* __restrict__ out, int total) {
  int t = blockIdx.x * blockDim.x + threadIdx.x;
  if (t >= total) return;
  int k = t % 1280, r = t / 1280, b = r >> 7;
  out[t] = (k < 256) ? l2pts[(size_t)r * 256 + k]
                     : (_Float16)l3[(size_t)b * 1024 + (k - 256)];
}

// ---------------------------------------------------------------------------
// 3-NN inverse-distance interpolation + concat staging (FP2/FP1)
// ---------------------------------------------------------------------------
__global__ void knn_interp_stage(const float* __restrict__ dxyz,
                                 const float* __restrict__ sxyz, int Nd, int Ns,
                                 const _Float16* __restrict__ pts1, int C1,
                                 const _Float16* __restrict__ spts, int C2,
                                 _Float16* __restrict__ out, int B) {
  int tid = blockIdx.x * blockDim.x + threadIdx.x;
  if (tid >= B * Nd) return;
  int b = tid / Nd;
  const float* p = dxyz + (size_t)tid * 3;
  float px = p[0], py = p[1], pz = p[2];
  const float* S = sxyz + (size_t)b * Ns * 3;
  float d0 = 1e30f, d1 = 1e30f, d2 = 1e30f;
  int i0 = 0, i1 = 0, i2 = 0;
  for (int j = 0; j < Ns; ++j) {
    float dx = S[j * 3 + 0] - px, dy = S[j * 3 + 1] - py, dz = S[j * 3 + 2] - pz;
    float dd = dx * dx + dy * dy + dz * dz;
    if (dd < d0)      { d2 = d1; i2 = i1; d1 = d0; i1 = i0; d0 = dd; i0 = j; }
    else if (dd < d1) { d2 = d1; i2 = i1; d1 = dd; i1 = j; }
    else if (dd < d2) { d2 = dd; i2 = j; }
  }
  float w0 = 1.f / (d0 + 1e-8f), w1 = 1.f / (d1 + 1e-8f), w2 = 1.f / (d2 + 1e-8f);
  float inv = 1.f / (w0 + w1 + w2);
  w0 *= inv; w1 *= inv; w2 *= inv;
  _Float16* o = out + (size_t)tid * (C1 + C2);
  if (C1 > 0) {
    const _Float16* p1 = pts1 + (size_t)tid * C1;
    for (int c = 0; c < C1; ++c) o[c] = p1[c];
  }
  const _Float16* s0 = spts + ((size_t)b * Ns + i0) * C2;
  const _Float16* s1 = spts + ((size_t)b * Ns + i1) * C2;
  const _Float16* s2 = spts + ((size_t)b * Ns + i2) * C2;
  for (int c = 0; c < C2; ++c)
    o[C1 + c] = (_Float16)(w0 * (float)s0[c] + w1 * (float)s1[c] + w2 * (float)s2[c]);
}

// ---------------------------------------------------------------------------
// Generic 2-layer row MLP (FP3/FP2): one wave = 32 rows. If the input rows fit
// (K0 <= 384), stage them into LDS with gfx1250 async-to-LDS DMA first.
// ---------------------------------------------------------------------------
__global__ void __launch_bounds__(32) fp_mlp2_kernel(
    const _Float16* __restrict__ in, int K0,
    const _Float16* w1, const float* b1, int N1,
    const _Float16* w2, const float* b2, int N2, _Float16* __restrict__ out) {
  __shared__ _Float16 bufIn[32 * 384];
  __shared__ _Float16 mid[32 * 256];
  int row0 = blockIdx.x * 32;
  int lane = threadIdx.x;
  const _Float16* src = in + (size_t)row0 * K0;
  if (K0 <= 384) {
    async_stage(src, bufIn, 32 * K0 * 2, lane);
    src = bufIn;
  }
  mlp_layer(src, K0, w1, b1, K0, N1, 2, mid, N1, lane);
  __syncthreads();
  mlp_layer(mid, N1, w2, b2, N1, N2, 2, out + (size_t)row0 * N2, N2, lane);
}

// ---------------------------------------------------------------------------
// FP1: 128 -> 128 -> 128 -> 128; input tile async-staged to LDS, final layer
// stores f32 to both output slots.
// ---------------------------------------------------------------------------
__global__ void __launch_bounds__(32) fp1_mlp_kernel(
    const _Float16* __restrict__ in,
    const _Float16* w1, const float* b1, const _Float16* w2, const float* b2,
    const _Float16* w3, const float* b3,
    float* __restrict__ outA, float* __restrict__ outB) {
  __shared__ _Float16 bufIn[32 * 128];
  __shared__ _Float16 bufA[32 * 128];
  __shared__ _Float16 bufB[32 * 128];
  int row0 = blockIdx.x * 32;
  int lane = threadIdx.x;
  async_stage(in + (size_t)row0 * 128, bufIn, 32 * 128 * 2, lane);
  mlp_layer(bufIn, 128, w1, b1, 128, 128, 2, bufA, 128, lane);
  __syncthreads();
  mlp_layer(bufA, 128, w2, b2, 128, 128, 2, bufB, 128, lane);
  __syncthreads();
  mlp_layer_f32dual(bufB, 128, w3, b3, 128, 128, 2,
                    outA + (size_t)row0 * 128, outB + (size_t)row0 * 128, 128, lane);
}

// ---------------------------------------------------------------------------
extern "C" void kernel_launch(void* const* d_in, const int* in_sizes, int n_in,
                              void* d_out, int out_size, void* d_ws, size_t ws_size,
                              hipStream_t stream) {
  (void)in_sizes; (void)n_in; (void)out_size; (void)ws_size;
  const int B = 16, N = 8192, S1 = 512, NS1 = 32, S2 = 128, NS2 = 64;

  const float* xyz = (const float*)d_in[0];
  const float *sa1w[3], *sa1b[3], *sa2w[3], *sa2b[3], *sa3w[3], *sa3b[3];
  for (int i = 0; i < 3; ++i) {
    sa1w[i] = (const float*)d_in[2 + 2 * i];  sa1b[i] = (const float*)d_in[3 + 2 * i];
    sa2w[i] = (const float*)d_in[8 + 2 * i];  sa2b[i] = (const float*)d_in[9 + 2 * i];
    sa3w[i] = (const float*)d_in[14 + 2 * i]; sa3b[i] = (const float*)d_in[15 + 2 * i];
  }
  const float* fp3w[2] = {(const float*)d_in[20], (const float*)d_in[22]};
  const float* fp3b[2] = {(const float*)d_in[21], (const float*)d_in[23]};
  const float* fp2w[2] = {(const float*)d_in[24], (const float*)d_in[26]};
  const float* fp2b[2] = {(const float*)d_in[25], (const float*)d_in[27]};
  const float* fp1w[3] = {(const float*)d_in[28], (const float*)d_in[30], (const float*)d_in[32]};
  const float* fp1b[3] = {(const float*)d_in[29], (const float*)d_in[31], (const float*)d_in[33]};

  char* ws = (char*)d_ws;
  size_t off = 0;
  auto alloc = [&](size_t bytes) -> void* {
    void* p = ws + off;
    off = (off + bytes + 255) & ~(size_t)255;
    return p;
  };
  auto cvt = [&](const float* src, int Nn, int Kk, int Kp) -> _Float16* {
    _Float16* dst = (_Float16*)alloc((size_t)Nn * Kp * sizeof(_Float16));
    int tot = Nn * Kp;
    cvt_w<<<(tot + 255) / 256, 256, 0, stream>>>(src, dst, Nn, Kk, Kp);
    return dst;
  };

  // f16 padded weights (live in L2 for the whole pipeline)
  _Float16* W11 = cvt(sa1w[0], 64, 6, 32);
  _Float16* W12 = cvt(sa1w[1], 64, 64, 64);
  _Float16* W13 = cvt(sa1w[2], 128, 64, 64);
  _Float16* W21 = cvt(sa2w[0], 128, 131, 160);
  _Float16* W22 = cvt(sa2w[1], 128, 128, 128);
  _Float16* W23 = cvt(sa2w[2], 256, 128, 128);
  _Float16* W31 = cvt(sa3w[0], 256, 259, 288);
  _Float16* W32 = cvt(sa3w[1], 512, 256, 256);
  _Float16* W33 = cvt(sa3w[2], 1024, 512, 512);
  _Float16* WF31 = cvt(fp3w[0], 256, 1280, 1280);
  _Float16* WF32 = cvt(fp3w[1], 256, 256, 256);
  _Float16* WF21 = cvt(fp2w[0], 256, 384, 384);
  _Float16* WF22 = cvt(fp2w[1], 128, 256, 256);
  _Float16* WF11 = cvt(fp1w[0], 128, 128, 128);
  _Float16* WF12 = cvt(fp1w[1], 128, 128, 128);
  _Float16* WF13 = cvt(fp1w[2], 128, 128, 128);

  // activation scratch
  int*      fps1   = (int*)alloc((size_t)B * S1 * 4);
  float*    l1xyz  = (float*)alloc((size_t)B * S1 * 3 * 4);
  int*      ball1  = (int*)alloc((size_t)B * S1 * NS1 * 4);
  _Float16* l1pts  = (_Float16*)alloc((size_t)B * S1 * 128 * 2);
  int*      fps2   = (int*)alloc((size_t)B * S2 * 4);
  float*    l2xyz  = (float*)alloc((size_t)B * S2 * 3 * 4);
  int*      ball2  = (int*)alloc((size_t)B * S2 * NS2 * 4);
  _Float16* l2pts  = (_Float16*)alloc((size_t)B * S2 * 256 * 2);
  float*    sa3par = (float*)alloc((size_t)B * 4 * 1024 * 4);
  float*    l3pts  = (float*)alloc((size_t)B * 1024 * 4);
  _Float16* fp3in  = (_Float16*)alloc((size_t)B * S2 * 1280 * 2);
  _Float16* fp3out = (_Float16*)alloc((size_t)B * S2 * 256 * 2);
  _Float16* fp2in  = (_Float16*)alloc((size_t)B * S1 * 384 * 2);
  _Float16* fp2out = (_Float16*)alloc((size_t)B * S1 * 128 * 2);
  _Float16* fp1in  = (_Float16*)alloc((size_t)B * N * 128 * 2);

  float* out_xyz = (float*)d_out;
  float* out_a   = out_xyz + (size_t)B * N * 3;
  float* out_b   = out_a + (size_t)B * N * 128;

  // ---- SA1 ----
  fps_kernel<16><<<B, 512, 0, stream>>>(xyz, N, S1, fps1, l1xyz);
  ball_query_kernel<<<B * S1, 32, 0, stream>>>(xyz, l1xyz, N, S1, 0.04f, NS1, ball1);
  sa1_kernel<<<B * S1, 32, 0, stream>>>(xyz, l1xyz, ball1,
      W11, sa1b[0], W12, sa1b[1], W13, sa1b[2], l1pts);

  // ---- SA2 ----
  fps_kernel<1><<<B, 512, 0, stream>>>(l1xyz, S1, S2, fps2, l2xyz);
  ball_query_kernel<<<B * S2, 32, 0, stream>>>(l1xyz, l2xyz, S1, S2, 0.16f, NS2, ball2);
  sa2_kernel<<<B * S2, 32, 0, stream>>>(l1xyz, l2xyz, ball2, l1pts,
      W21, sa2b[0], W22, sa2b[1], W23, sa2b[2], l2pts);

  // ---- SA3 (global) ----
  sa3_kernel<<<B * 4, 32, 0, stream>>>(l2xyz, l2pts,
      W31, sa3b[0], W32, sa3b[1], W33, sa3b[2], sa3par);
  sa3_combine<<<(B * 1024 + 255) / 256, 256, 0, stream>>>(sa3par, l3pts);

  // ---- FP3: broadcast interp + concat + MLP 1280->256->256 ----
  {
    int tot = B * S2 * 1280;
    fp3_stage<<<(tot + 255) / 256, 256, 0, stream>>>(l2pts, l3pts, fp3in, tot);
  }
  fp_mlp2_kernel<<<B * S2 / 32, 32, 0, stream>>>(fp3in, 1280,
      WF31, fp3b[0], 256, WF32, fp3b[1], 256, fp3out);

  // ---- FP2: 3-NN interp + concat[l1_pts, interp] + MLP 384->256->128 ----
  knn_interp_stage<<<(B * S1 + 255) / 256, 256, 0, stream>>>(
      l1xyz, l2xyz, S1, S2, l1pts, 128, fp3out, 256, fp2in, B);
  fp_mlp2_kernel<<<B * S1 / 32, 32, 0, stream>>>(fp2in, 384,
      WF21, fp2b[0], 256, WF22, fp2b[1], 128, fp2out);

  // ---- FP1: 3-NN interp (no points1) + MLP 128->128->128->128 ----
  knn_interp_stage<<<(B * N + 255) / 256, 256, 0, stream>>>(
      xyz, l1xyz, N, S1, (const _Float16*)nullptr, 0, fp2out, 128, fp1in, B);
  fp1_mlp_kernel<<<B * N / 32, 32, 0, stream>>>(fp1in,
      WF11, fp1b[0], WF12, fp1b[1], WF13, fp1b[2], out_a, out_b);

  // ---- output: xyz pass-through ----
  {
    int tot = B * N * 3;
    copy_f32<<<(tot + 255) / 256, 256, 0, stream>>>(xyz, out_xyz, tot);
  }
}
